// MapBasedMultiHeadAttention_2302102471367
// MI455X (gfx1250) — compile-verified, matrix operand order checked by reading, not run on hardware
//
#include <hip/hip_runtime.h>
#include <hip/hip_bf16.h>
#include <math.h>

// ---------------------------------------------------------------------------
// MapBasedMultiHeadAttention for MI455X (gfx1250, wave32, WMMA)
// GEMMs on v_wmma_f32_16x16x32_bf16 with pre-converted bf16 operands so the
// inner loops are pure b128-load + WMMA (no per-iteration cvt VALU).
// ---------------------------------------------------------------------------

typedef __attribute__((ext_vector_type(16))) __bf16 v16bf;
typedef __attribute__((ext_vector_type(8)))  float  v8f;
typedef unsigned short u16;
typedef unsigned int   u32;

#define D_MODEL 512
#define N_HEAD  8
#define DK      64
#define BATCH   4
#define LQ      1024
#define LK      1024
#define ROWS    (BATCH * LQ)                 // 4096
#define OUT0    ((size_t)ROWS * D_MODEL)     // 2,097,152 floats (out tensor)
#define SCORES  (BATCH * N_HEAD * LQ)        // 32768

__device__ __forceinline__ u16 f2bf(float f) {
  u32 x = __float_as_uint(f);
  x = (x + 0x7fffu + ((x >> 16) & 1u)) >> 16;   // round-to-nearest-even
  return (u16)x;
}
__device__ __forceinline__ u32 pack2(float lo, float hi) {
  return (u32)f2bf(lo) | ((u32)f2bf(hi) << 16);
}

__device__ __forceinline__ v8f wmma_bf16(v16bf a, v16bf b, v8f c) {
  return __builtin_amdgcn_wmma_f32_16x16x32_bf16(false, a, false, b, (short)0, c,
                                                 false, false);
}

// bf16 row-major fragment: lane r holds row/col r, K {0..7,16..23} (half=0)
// or K {8..15,24..31} (half=1).  Two b128 loads, zero repacking.
__device__ __forceinline__ v16bf load_frag_bf16(const u16* __restrict__ base,
                                                int ld, int r, int half, int k0) {
  const u16* p = base + (size_t)r * ld + k0 + half * 8;
  union { v16bf v; uint4 q[2]; } f;
  f.q[0] = *(const uint4*)(p);
  f.q[1] = *(const uint4*)(p + 16);
  return f.v;
}

// fp32 row-major fragment with inline bf16 conversion (used only for attn).
__device__ __forceinline__ v16bf load_frag_f32(const float* __restrict__ base,
                                               int ld, int r, int half, int k0) {
  const float* p = base + (size_t)r * ld + k0 + half * 8;
  const float4 x0 = *(const float4*)(p + 0);
  const float4 x1 = *(const float4*)(p + 4);
  const float4 x2 = *(const float4*)(p + 16);
  const float4 x3 = *(const float4*)(p + 20);
  union { v16bf v; uint4 q[2]; } f;
  f.q[0] = make_uint4(pack2(x0.x, x0.y), pack2(x0.z, x0.w),
                      pack2(x1.x, x1.y), pack2(x1.z, x1.w));
  f.q[1] = make_uint4(pack2(x2.x, x2.y), pack2(x2.z, x2.w),
                      pack2(x3.x, x3.y), pack2(x3.z, x3.w));
  return f.v;
}

// ---------------- K0: fp32 -> bf16 bulk convert (8 elems / thread) ----------
__global__ void cvt_kernel(const float* __restrict__ src, uint4* __restrict__ dst,
                           int n8) {
  int gid = blockIdx.x * blockDim.x + threadIdx.x;
  if (gid >= n8) return;
  const float4* s4 = (const float4*)src;
  float4 a = s4[2 * gid], b = s4[2 * gid + 1];
  dst[gid] = make_uint4(pack2(a.x, a.y), pack2(a.z, a.w),
                        pack2(b.x, b.y), pack2(b.z, b.w));
}

// ---------------- K1: Q/K/V projections -------------------------------------
// One wave computes a 16x64 strip (4 accumulators, A fragment reused 4x).
// mode 0/1 -> fp32 qp/kp (needed for score dots); mode 2 -> transposed bf16
// vpt [b,h,dv,lk] so attn@V gets K-contiguous B fragments.
__global__ void proj_kernel(const u16* __restrict__ qb, const u16* __restrict__ kb,
                            const u16* __restrict__ vb,
                            const u16* __restrict__ wqb, const u16* __restrict__ wkb,
                            const u16* __restrict__ wvb,
                            float* __restrict__ qp, float* __restrict__ kp,
                            u16* __restrict__ vpt) {
  const u16* A; const u16* W;
  int mode = blockIdx.y;
  if (mode == 0)      { A = qb; W = wqb; }
  else if (mode == 1) { A = kb; W = wkb; }
  else                { A = vb; W = wvb; }

  int wave = (blockIdx.x * blockDim.x + threadIdx.x) >> 5;   // 2048 waves
  int lane = threadIdx.x & 31;
  int r = lane & 15, half = lane >> 4;
  int row0 = (wave >> 3) << 4;       // 256 row tiles
  int col0 = (wave & 7) << 6;        // 8 blocks of 64 cols

  const u16* Abase = A + (size_t)row0 * D_MODEL;
  const u16* W0 = W + (size_t)(col0 +  0) * D_MODEL;
  const u16* W1 = W + (size_t)(col0 + 16) * D_MODEL;
  const u16* W2 = W + (size_t)(col0 + 32) * D_MODEL;
  const u16* W3 = W + (size_t)(col0 + 48) * D_MODEL;

  v8f acc[4] = {};
  for (int k0 = 0; k0 < D_MODEL; k0 += 32) {
    __builtin_prefetch(Abase + (size_t)r * D_MODEL + k0 + 64, 0, 1);
    v16bf a  = load_frag_bf16(Abase, D_MODEL, r, half, k0);
    v16bf b0 = load_frag_bf16(W0, D_MODEL, r, half, k0);
    v16bf b1 = load_frag_bf16(W1, D_MODEL, r, half, k0);
    v16bf b2 = load_frag_bf16(W2, D_MODEL, r, half, k0);
    v16bf b3 = load_frag_bf16(W3, D_MODEL, r, half, k0);
    acc[0] = wmma_bf16(a, b0, acc[0]);
    acc[1] = wmma_bf16(a, b1, acc[1]);
    acc[2] = wmma_bf16(a, b2, acc[2]);
    acc[3] = wmma_bf16(a, b3, acc[3]);
  }

  if (mode < 2) {
    float* C = (mode == 0) ? qp : kp;
#pragma unroll
    for (int t = 0; t < 4; ++t) {
      float* Crow = C + (size_t)(row0 + half * 8) * D_MODEL + col0 + t * 16 + r;
#pragma unroll
      for (int vv = 0; vv < 8; ++vv)
        Crow[(size_t)vv * D_MODEL] = acc[t][vv];
    }
  } else {
    int b = row0 >> 10;
    int j = (row0 & 1023) + half * 8;   // 8 consecutive lk positions
#pragma unroll
    for (int t = 0; t < 4; ++t) {
      int col = col0 + t * 16 + r;      // global col in [0,512)
      int h = col >> 6, n = col & 63;
      u16* dst = vpt + ((size_t)((b * N_HEAD + h) * DK + n)) * LK + j;
      *(uint4*)dst = make_uint4(pack2(acc[t][0], acc[t][1]),
                                pack2(acc[t][2], acc[t][3]),
                                pack2(acc[t][4], acc[t][5]),
                                pack2(acc[t][6], acc[t][7]));
    }
  }
}

// ---------------- K2: additive scores sq/sk (64-wide dots) ------------------
__global__ void score_kernel(const float* __restrict__ qp, const float* __restrict__ kp,
                             const float* __restrict__ wm,
                             float* __restrict__ sq, float* __restrict__ sk) {
  int gid = blockIdx.x * blockDim.x + threadIdx.x;     // 65536 threads
  int isK = gid >= SCORES;
  int id  = gid & (SCORES - 1);
  int i   = id & 1023;
  int bh  = id >> 10;              // b*8 + h
  int b   = bh >> 3, h = bh & 7;
  const float* src = isK ? kp : qp;
  const float* w   = wm + (isK ? DK : 0);
  const float* row = src + ((size_t)(b * LQ + i)) * D_MODEL + h * DK;
  float s = 0.f;
#pragma unroll
  for (int d = 0; d < DK; ++d) s += row[d] * w[d];
  (isK ? sk : sq)[id] = s;
}

// ---------------- K3: softmax rows -> attn (head-major [h*b, lq, lk]) -------
__global__ void softmax_kernel(const float* __restrict__ sq, const float* __restrict__ sk,
                               const unsigned char* __restrict__ mask,
                               float* __restrict__ attn) {
  __shared__ float red[256];
  int rid = blockIdx.x;                 // 32768 rows
  int tid = threadIdx.x;                // 256
  int i  = rid & 1023;
  int hb = rid >> 10;                   // h*4 + b
  int h  = hb >> 2, b = hb & 3;
  int bh = (b << 3) | h;

  float si = sq[bh * LQ + i];
  const float* skr = sk + bh * LK;
  const unsigned char* mr = mask + ((size_t)b * LQ + i) * LK;
  float* ar = attn + (size_t)rid * LK;

  float vals[4];
  float mx = -3.4e38f;
#pragma unroll
  for (int t = 0; t < 4; ++t) {
    int j = tid + (t << 8);
    float s = si + skr[j];
    if (mr[j]) s = -1e10f;
    vals[t] = s;
    mx = fmaxf(mx, s);
  }
  red[tid] = mx; __syncthreads();
  for (int s = 128; s > 0; s >>= 1) {
    if (tid < s) red[tid] = fmaxf(red[tid], red[tid + s]);
    __syncthreads();
  }
  mx = red[0]; __syncthreads();

  float sum = 0.f;
#pragma unroll
  for (int t = 0; t < 4; ++t) { vals[t] = __expf(vals[t] - mx); sum += vals[t]; }
  red[tid] = sum; __syncthreads();
  for (int s = 128; s > 0; s >>= 1) {
    if (tid < s) red[tid] += red[tid + s];
    __syncthreads();
  }
  float inv = 1.0f / red[0];
#pragma unroll
  for (int t = 0; t < 4; ++t) ar[tid + (t << 8)] = vals[t] * inv;
}

// ---------------- K4: attn @ V ---------------------------------------------
// One wave computes 16 rows x full dv=64 for one (h,b).  A: fp32 attn with
// inline cvt; B: transposed bf16 vpt (K-contiguous, two b128 loads/frag).
// Output written directly as bf16 (zb) for the fc GEMM.
__global__ void attnv_kernel(const float* __restrict__ attn, const u16* __restrict__ vpt,
                             u16* __restrict__ zb) {
  int wave = (blockIdx.x * blockDim.x + threadIdx.x) >> 5;  // 2048 waves
  int lane = threadIdx.x & 31;
  int r = lane & 15, half = lane >> 4;
  int hb = wave >> 6;                 // h*4 + b
  int ti = wave & 63;                 // 64 row tiles
  int h = hb >> 2, b = hb & 3;

  const float* abase = attn + ((size_t)hb * LQ + (size_t)ti * 16) * LK;
  const u16*   vbase = vpt + ((size_t)((b * N_HEAD + h) * DK)) * LK;

  v8f acc[4] = {};
  for (int j0 = 0; j0 < LK; j0 += 32) {
    __builtin_prefetch(abase + (size_t)r * LK + j0 + 64, 0, 1);
    v16bf a = load_frag_f32(abase, LK, r, half, j0);
#pragma unroll
    for (int t = 0; t < 4; ++t) {
      v16bf bf = load_frag_bf16(vbase, LK, t * 16 + r, half, j0);
      acc[t] = wmma_bf16(a, bf, acc[t]);
    }
  }
#pragma unroll
  for (int t = 0; t < 4; ++t) {
    int col = h * DK + t * 16 + r;
#pragma unroll
    for (int vv = 0; vv < 8; ++vv) {
      size_t row = (size_t)b * LQ + (size_t)ti * 16 + half * 8 + vv;
      zb[row * D_MODEL + col] = f2bf(acc[t][vv]);
    }
  }
}

// ---------------- K5a: fc GEMM + bias + LeakyReLU + residual ---------------
__global__ void fc_kernel(const u16* __restrict__ zb, const u16* __restrict__ fwb,
                          const float* __restrict__ fcb, const float* __restrict__ resid,
                          float* __restrict__ t) {
  int wave = (blockIdx.x * blockDim.x + threadIdx.x) >> 5;   // 2048 waves
  int lane = threadIdx.x & 31;
  int r = lane & 15, half = lane >> 4;
  int row0 = (wave >> 3) << 4;
  int col0 = (wave & 7) << 6;

  const u16* Abase = zb + (size_t)row0 * D_MODEL;
  const u16* W0 = fwb + (size_t)(col0 +  0) * D_MODEL;
  const u16* W1 = fwb + (size_t)(col0 + 16) * D_MODEL;
  const u16* W2 = fwb + (size_t)(col0 + 32) * D_MODEL;
  const u16* W3 = fwb + (size_t)(col0 + 48) * D_MODEL;

  v8f acc[4] = {};
  for (int k0 = 0; k0 < D_MODEL; k0 += 32) {
    __builtin_prefetch(Abase + (size_t)r * D_MODEL + k0 + 64, 0, 1);
    v16bf a  = load_frag_bf16(Abase, D_MODEL, r, half, k0);
    v16bf b0 = load_frag_bf16(W0, D_MODEL, r, half, k0);
    v16bf b1 = load_frag_bf16(W1, D_MODEL, r, half, k0);
    v16bf b2 = load_frag_bf16(W2, D_MODEL, r, half, k0);
    v16bf b3 = load_frag_bf16(W3, D_MODEL, r, half, k0);
    acc[0] = wmma_bf16(a, b0, acc[0]);
    acc[1] = wmma_bf16(a, b1, acc[1]);
    acc[2] = wmma_bf16(a, b2, acc[2]);
    acc[3] = wmma_bf16(a, b3, acc[3]);
  }
#pragma unroll
  for (int tt = 0; tt < 4; ++tt) {
    int col = col0 + tt * 16 + r;
    float bias = fcb[col];
#pragma unroll
    for (int vv = 0; vv < 8; ++vv) {
      int row = row0 + half * 8 + vv;
      float val = acc[tt][vv] + bias;
      val = (val >= 0.f) ? val : 0.2f * val;           // LeakyReLU(0.2)
      val += resid[(size_t)row * D_MODEL + col];       // residual = original q
      t[(size_t)row * D_MODEL + col] = val;
    }
  }
}

// ---------------- K5b: LayerNorm over d_model=512 --------------------------
__global__ void ln_kernel(const float* __restrict__ t, const float* __restrict__ g,
                          const float* __restrict__ be, float* __restrict__ out) {
  __shared__ float s1[256];
  __shared__ float s2[256];
  int m = blockIdx.x, tid = threadIdx.x;
  const float* rowp = t + (size_t)m * D_MODEL;
  float x0 = rowp[tid], x1 = rowp[tid + 256];
  s1[tid] = x0 + x1;
  s2[tid] = x0 * x0 + x1 * x1;
  __syncthreads();
  for (int s = 128; s > 0; s >>= 1) {
    if (tid < s) { s1[tid] += s1[tid + s]; s2[tid] += s2[tid + s]; }
    __syncthreads();
  }
  float mu  = s1[0] * (1.0f / (float)D_MODEL);
  float var = s2[0] * (1.0f / (float)D_MODEL) - mu * mu;
  float inv = rsqrtf(var + 1e-5f);
  out[(size_t)m * D_MODEL + tid]       = (x0 - mu) * inv * g[tid]       + be[tid];
  out[(size_t)m * D_MODEL + tid + 256] = (x1 - mu) * inv * g[tid + 256] + be[tid + 256];
}

// ---------------------------------------------------------------------------
extern "C" void kernel_launch(void* const* d_in, const int* in_sizes, int n_in,
                              void* d_out, int out_size, void* d_ws, size_t ws_size,
                              hipStream_t stream) {
  const float* q    = (const float*)d_in[0];
  const float* k    = (const float*)d_in[1];
  const float* v    = (const float*)d_in[2];
  const unsigned char* mask = (const unsigned char*)d_in[3];
  const float* Wq   = (const float*)d_in[4];
  const float* Wk   = (const float*)d_in[5];
  const float* Wv   = (const float*)d_in[6];
  const float* wm   = (const float*)d_in[7];
  const float* fcw  = (const float*)d_in[8];
  const float* fcb  = (const float*)d_in[9];
  const float* lng  = (const float*)d_in[10];
  const float* lnb  = (const float*)d_in[11];

  float* out  = (float*)d_out;
  float* attn = out + OUT0;                 // attn_flat [32,1024,1024]

  // -------- workspace carving (256B aligned blocks) --------
  char* cur = (char*)d_ws;
  auto carve = [&cur](size_t bytes) -> char* {
    char* p = cur;
    cur += (bytes + 255) & ~(size_t)255;
    return p;
  };
  const size_t NE_ACT = (size_t)ROWS * D_MODEL;     // 2,097,152
  const size_t NE_W   = (size_t)D_MODEL * D_MODEL;  // 262,144

  u16*   qb  = (u16*)carve(NE_ACT * 2);
  u16*   kb  = (u16*)carve(NE_ACT * 2);
  u16*   vb  = (u16*)carve(NE_ACT * 2);
  u16*   wqb = (u16*)carve(NE_W * 2);
  u16*   wkb = (u16*)carve(NE_W * 2);
  u16*   wvb = (u16*)carve(NE_W * 2);
  u16*   fwb = (u16*)carve(NE_W * 2);
  float* qp  = (float*)carve(NE_ACT * 4);
  float* kp  = (float*)carve(NE_ACT * 4);
  u16*   vpt = (u16*)carve(NE_ACT * 2);             // [b,h,dv,lk]
  float* sq  = (float*)carve(SCORES * 4);
  float* sk  = (float*)carve(SCORES * 4);
  u16*   zb  = (u16*)carve(NE_ACT * 2);             // attn@V, bf16 row-major
  float* t   = (float*)carve(NE_ACT * 4);           // pre-LN fp32

  // K0: bulk bf16 conversion
  cvt_kernel<<<(NE_ACT / 8 + 255) / 256, 256, 0, stream>>>(q,   (uint4*)qb,  NE_ACT / 8);
  cvt_kernel<<<(NE_ACT / 8 + 255) / 256, 256, 0, stream>>>(k,   (uint4*)kb,  NE_ACT / 8);
  cvt_kernel<<<(NE_ACT / 8 + 255) / 256, 256, 0, stream>>>(v,   (uint4*)vb,  NE_ACT / 8);
  cvt_kernel<<<(NE_W   / 8 + 255) / 256, 256, 0, stream>>>(Wq,  (uint4*)wqb, NE_W / 8);
  cvt_kernel<<<(NE_W   / 8 + 255) / 256, 256, 0, stream>>>(Wk,  (uint4*)wkb, NE_W / 8);
  cvt_kernel<<<(NE_W   / 8 + 255) / 256, 256, 0, stream>>>(Wv,  (uint4*)wvb, NE_W / 8);
  cvt_kernel<<<(NE_W   / 8 + 255) / 256, 256, 0, stream>>>(fcw, (uint4*)fwb, NE_W / 8);

  dim3 pg(512, 3, 1);                       // 2048 waves per projection
  proj_kernel<<<pg, 128, 0, stream>>>(qb, kb, vb, wqb, wkb, wvb, qp, kp, vpt);
  score_kernel<<<256, 256, 0, stream>>>(qp, kp, wm, sq, sk);
  softmax_kernel<<<32768, 256, 0, stream>>>(sq, sk, mask, attn);
  attnv_kernel<<<512, 128, 0, stream>>>(attn, vpt, zb);
  fc_kernel<<<512, 128, 0, stream>>>(zb, fwb, fcb, q, t);
  ln_kernel<<<4096, 256, 0, stream>>>(t, lng, lnb, out);
}
